// self_Attention_1D_for_timestep_position_63127429317276
// MI455X (gfx1250) — matrix operand order repaired and checked by reading, not compiled, not run on hardware
//
#include <hip/hip_runtime.h>

typedef __attribute__((ext_vector_type(16))) _Float16 v16h;
typedef __attribute__((ext_vector_type(8)))  _Float16 v8h;
typedef __attribute__((ext_vector_type(8)))  float    v8f;

#define B_ 4
#define C_ 64
#define L_ 4096
#define BN_EPS 1e-5f

// ---------------------------------------------------------------------------
// Kernel 1: x = input + pos_embed ; q,k,v = BN(ReLU(W x + b)).
// Writes: xbuf f32 [B][C][L] (residual), qT/kT f16 [B][L][C], vC f16 [B][C][L]
// pos-embed (faithful to int-cast): ch0=sin(l), ch1=cos(l), even>=2:0, odd>=3:1
// ---------------------------------------------------------------------------
__launch_bounds__(256)
__global__ void prep_kernel(const float* __restrict__ in,
                            const float* __restrict__ Wq, const float* __restrict__ bq,
                            const float* __restrict__ Wk, const float* __restrict__ bk,
                            const float* __restrict__ Wv, const float* __restrict__ bv,
                            const float* __restrict__ bnw, const float* __restrict__ bnb,
                            float* __restrict__ xbuf,
                            _Float16* __restrict__ qT, _Float16* __restrict__ kT,
                            _Float16* __restrict__ vC)
{
    __shared__ float sW[3][C_ * C_];
    const int tid = threadIdx.x;
    for (int i = tid; i < C_ * C_; i += blockDim.x) {
        sW[0][i] = Wq[i];
        sW[1][i] = Wk[i];
        sW[2][i] = Wv[i];
    }
    __syncthreads();

    // 4 threads per (b,l): each computes a 16-channel output chunk.
    const int gid = blockIdx.x * blockDim.x + tid;   // 0 .. B*L*4-1
    const int oc  = gid & 3;
    const int pos = gid >> 2;
    const int b   = pos / L_;
    const int l   = pos % L_;

    const float fl = (float)l;
    float x[C_];
    #pragma unroll
    for (int c = 0; c < C_; ++c) {
        float pe = (c == 0) ? sinf(fl)
                 : (c == 1) ? cosf(fl)
                 : ((c & 1) ? 1.0f : 0.0f);
        float xv = in[((size_t)b * C_ + c) * L_ + l] + pe;
        x[c] = xv;
        if (oc == 0) xbuf[((size_t)b * C_ + c) * L_ + l] = xv;
    }

    const float inv_s = rsqrtf(1.0f + BN_EPS);
    const int o0 = oc * 16;
    for (int oo = 0; oo < 16; ++oo) {
        int o = o0 + oo;
        float aq = bq[o], ak = bk[o], av = bv[o];
        #pragma unroll 8
        for (int c = 0; c < C_; ++c) {
            float xv = x[c];
            aq = fmaf(sW[0][o * C_ + c], xv, aq);
            ak = fmaf(sW[1][o * C_ + c], xv, ak);
            av = fmaf(sW[2][o * C_ + c], xv, av);
        }
        const float sc = bnw[o] * inv_s, bb = bnb[o];
        aq = (aq > 0.0f ? aq : 0.0f) * sc + bb;
        ak = (ak > 0.0f ? ak : 0.0f) * sc + bb;
        av = (av > 0.0f ? av : 0.0f) * sc + bb;
        qT[((size_t)b * L_ + l) * C_ + o] = (_Float16)aq;
        kT[((size_t)b * L_ + l) * C_ + o] = (_Float16)ak;
        vC[((size_t)b * C_ + o) * L_ + l] = (_Float16)av;
    }
}

// ---------------------------------------------------------------------------
// Kernel 2: one workgroup (8 wave32) per (b, 16-row strip of L).
// LDS: Sbuf f32[16][4096] (256 KB) + stats + per-wave f16 staging tiles.
// Phase 1: scores via v_wmma_f32_16x16x32_f16 (K=64, 2 independent accums)
// Phase 2: row max, exp-in-place, row sums
// Phase 3: normalize, write attn (268 MB = HBM floor), out += P*v^T via WMMA
// ---------------------------------------------------------------------------
#define NWAVE 8
#define LDS_SBUF_BYTES (16 * L_ * 4)                 // 262144
#define LDS_STATS      (LDS_SBUF_BYTES)              // rowmax[16], rowinv[16]
#define LDS_REDSC      (LDS_SBUF_BYTES + 128)        // 256 f32
#define LDS_STAGE      (LDS_SBUF_BYTES + 128 + 1024) // NWAVE * 16*32 f16
#define LDS_TOTAL      (LDS_STAGE + NWAVE * 16 * 32 * 2)

__launch_bounds__(32 * NWAVE, 1)
__global__ void attn_kernel(const _Float16* __restrict__ qT,
                            const _Float16* __restrict__ kT,
                            const _Float16* __restrict__ vC,
                            const float* __restrict__ xbuf,
                            const float* __restrict__ gamma_p,
                            float* __restrict__ out,    // [B][C][L]
                            float* __restrict__ attn)   // [B][L][L]
{
    extern __shared__ char smem[];
    float*    Sbuf   = (float*)smem;                       // [16][4096]
    float*    rowmax = (float*)(smem + LDS_STATS);         // [16]
    float*    rowinv = rowmax + 16;                        // [16]
    float*    redsc  = (float*)(smem + LDS_REDSC);         // [256]
    _Float16* stage  = (_Float16*)(smem + LDS_STAGE);      // [NWAVE][16][32]

    const int  tid  = threadIdx.x;
    const int  wid  = tid >> 5;
    const int  lane = tid & 31;
    const int  lo16 = lane & 15;
    const bool hi   = lane >= 16;

    const int b  = blockIdx.y;
    const int m0 = blockIdx.x * 16;   // row strip base

    // ---- A-tiles of q^T for this strip (M=l rows, K=c), loaded once ----
    // A layout: lanes 0-15 M=row, K 0..15; lanes 16-31 K 16..31
    v16h aq[2];
    {
        const _Float16* qrow = qT + ((size_t)b * L_ + m0 + lo16) * C_;
        #pragma unroll
        for (int kc = 0; kc < 2; ++kc) {
            const _Float16* p = qrow + kc * 32 + (hi ? 16 : 0);
            v8h c0 = *(const v8h*)(p);
            v8h c1 = *(const v8h*)(p + 8);
            #pragma unroll
            for (int i = 0; i < 8; ++i) { aq[kc][i] = c0[i]; aq[kc][i + 8] = c1[i]; }
        }
    }

    // ---- Phase 1: scores strip -> LDS (independent accumulators, no RAW) ----
    for (int j = wid; j < L_ / 16; j += NWAVE) {
        // B layout: lanes 0-15 N=col (K 0..15), lanes 16-31 N=col (K 16..31)
        const _Float16* krow = kT + ((size_t)b * L_ + j * 16 + lo16) * C_;
        v16h b0 = *(const v16h*)(krow + (hi ? 16 : 0));
        v16h b1 = *(const v16h*)(krow + 32 + (hi ? 16 : 0));
        v8f z = {};
        v8f acc0 = __builtin_amdgcn_wmma_f32_16x16x32_f16(
            false, aq[0], false, b0, (short)0, z, false, false);
        v8f acc1 = __builtin_amdgcn_wmma_f32_16x16x32_f16(
            false, aq[1], false, b1, (short)0, z, false, false);
        const float scale = 0.015625f;   // 1/sqrt(4096)
        const int col = j * 16 + lo16;
        #pragma unroll
        for (int r = 0; r < 8; ++r) {
            int row = r + (hi ? 8 : 0);
            Sbuf[row * L_ + col] = (acc0[r] + acc1[r]) * scale;
        }
    }
    __syncthreads();

    // ---- Phase 2: softmax statistics (16 threads per row) ----
    {
        const int row = tid >> 4, seg = tid & 15;
        float* srow = Sbuf + row * L_ + seg * 256;
        float m = -3.0e38f;
        for (int i = 0; i < 256; ++i) {
            float s = srow[i];
            m = (s > m) ? s : m;           // finite scores: plain select
        }
        redsc[tid] = m;
        __syncthreads();
        if (tid < 16) {
            float mm = redsc[tid * 16];
            #pragma unroll
            for (int i = 1; i < 16; ++i) {
                float s = redsc[tid * 16 + i];
                mm = (s > mm) ? s : mm;
            }
            rowmax[tid] = mm;
        }
        __syncthreads();
        const float rm = rowmax[row];
        float s = 0.0f;
        for (int i = 0; i < 256; ++i) {
            float e = __expf(srow[i] - rm);
            srow[i] = e;                   // exp in place
            s += e;
        }
        redsc[tid] = s;
        __syncthreads();
        if (tid < 16) {
            float ss = 0.0f;
            #pragma unroll
            for (int i = 0; i < 16; ++i) ss += redsc[tid * 16 + i];
            rowinv[tid] = 1.0f / ss;
        }
        __syncthreads();
    }

    float rinv[8];
    #pragma unroll
    for (int r = 0; r < 8; ++r) rinv[r] = rowinv[r + (hi ? 8 : 0)];

    // ---- Phase 3: write attn + out += P * v^T ----
    v8f oacc[4] = {};
    _Float16* mystage = stage + wid * 16 * 32;    // wave-private 16x32 f16

    for (int p = wid; p < 128; p += NWAVE) {      // 16 K-chunks of 32 per wave
        #pragma unroll
        for (int t = 0; t < 2; ++t) {
            const int col = (p * 2 + t) * 16 + lo16;
            #pragma unroll
            for (int r = 0; r < 8; ++r) {
                int row = r + (hi ? 8 : 0);
                float pv = Sbuf[row * L_ + col] * rinv[r];
                attn[((size_t)b * L_ + m0 + row) * L_ + col] = pv;
                mystage[row * 32 + t * 16 + lo16] = (_Float16)pv;   // C/D -> A relayout
            }
        }
        __syncthreads();   // uniform trip count for all waves

        v16h ap;
        {
            const _Float16* sp = mystage + lo16 * 32 + (hi ? 16 : 0);
            v8h c0 = *(const v8h*)(sp);
            v8h c1 = *(const v8h*)(sp + 8);
            #pragma unroll
            for (int i = 0; i < 8; ++i) { ap[i] = c0[i]; ap[i + 8] = c1[i]; }
        }
        const int mk = p * 32 + (hi ? 16 : 0);
        #pragma unroll
        for (int cn = 0; cn < 4; ++cn) {      // 4 independent WMMA chains
            v16h bv_ = *(const v16h*)(vC + ((size_t)b * C_ + cn * 16 + lo16) * L_ + mk);
            oacc[cn] = __builtin_amdgcn_wmma_f32_16x16x32_f16(
                false, ap, false, bv_, (short)0, oacc[cn], false, false);
        }
    }
    __syncthreads();

    // ---- cross-wave reduction (reuse Sbuf) + residual + gamma ----
    float* red = Sbuf;                       // [NWAVE][16][64]
    #pragma unroll
    for (int cn = 0; cn < 4; ++cn)
        #pragma unroll
        for (int r = 0; r < 8; ++r) {
            int row = r + (hi ? 8 : 0);
            red[(wid * 16 + row) * 64 + cn * 16 + lo16] = oacc[cn][r];
        }
    __syncthreads();

    const float g = gamma_p[0];
    for (int idx = tid; idx < 16 * 64; idx += 32 * NWAVE) {
        float v = 0.0f;
        #pragma unroll
        for (int w = 0; w < NWAVE; ++w) v += red[w * 1024 + idx];
        int row = idx >> 6, c = idx & 63;
        size_t off = ((size_t)b * C_ + c) * L_ + (m0 + row);
        out[off] = g * v + xbuf[off];
    }
}

// ---------------------------------------------------------------------------
extern "C" void kernel_launch(void* const* d_in, const int* in_sizes, int n_in,
                              void* d_out, int out_size, void* d_ws, size_t ws_size,
                              hipStream_t stream)
{
    const float* in    = (const float*)d_in[0];
    const float* Wq    = (const float*)d_in[1];
    const float* bq    = (const float*)d_in[2];
    const float* Wk    = (const float*)d_in[3];
    const float* bk    = (const float*)d_in[4];
    const float* Wv    = (const float*)d_in[5];
    const float* bv    = (const float*)d_in[6];
    const float* bnw   = (const float*)d_in[7];
    const float* bnb   = (const float*)d_in[8];
    const float* gamma = (const float*)d_in[9];

    char* ws = (char*)d_ws;
    float*    xbuf = (float*)ws;                                  // 4 MiB
    _Float16* qT   = (_Float16*)(ws + (size_t)4 * 1024 * 1024);   // 2 MiB
    _Float16* kT   = (_Float16*)(ws + (size_t)6 * 1024 * 1024);   // 2 MiB
    _Float16* vC   = (_Float16*)(ws + (size_t)8 * 1024 * 1024);   // 2 MiB

    float* out  = (float*)d_out;                   // [B][C][L]
    float* attn = out + (size_t)B_ * C_ * L_;      // [B][L][L]

    // Allow >64KB dynamic LDS (hardware supports 320KB/WG). Host-side call,
    // executes immediately (not a stream op), safe under graph capture.
    (void)hipFuncSetAttribute((const void*)attn_kernel,
                              hipFuncAttributeMaxDynamicSharedMemorySize,
                              LDS_TOTAL);

    prep_kernel<<<dim3((B_ * L_ * 4) / 256), 256, 0, stream>>>(
        in, Wq, bq, Wk, bk, Wv, bv, bnw, bnb, xbuf, qT, kT, vC);

    attn_kernel<<<dim3(L_ / 16, B_), 32 * NWAVE, LDS_TOTAL, stream>>>(
        qT, kT, vC, xbuf, gamma, out, attn);
}
